// FixedProductionSplatFlowAttention_60206851555463
// MI455X (gfx1250) — matrix operand order repaired and checked by reading, not compiled, not run on hardware
//
#include <hip/hip_runtime.h>
#include <hip/hip_bf16.h>

typedef __attribute__((ext_vector_type(16))) _Float16 v16h;
typedef __attribute__((ext_vector_type(8)))  float    v8f;

#define B_ 4
#define T_ 2048
#define D_ 1024
#define S_ 64
#define EPS_ 1e-8f

// 16-byte POD chunk of 8 halves for vector loads of A-operand halves
struct alignas(16) H8 { _Float16 h[8]; };
union AV { _Float16 h[16]; v16h v; };

// ---- CDNA5 async global->LDS path (ASYNCcnt-tracked), with guarded fallback ----
#if __has_builtin(__builtin_amdgcn_global_load_async_to_lds_b128)
#define HAVE_ASYNC_LDS 1
#else
#define HAVE_ASYNC_LDS 0
#endif

#if __has_builtin(__builtin_amdgcn_s_wait_asynccnt)
#define ASYNC_WAIT(n) __builtin_amdgcn_s_wait_asynccnt(n)
#else
#define ASYNC_WAIT(n) asm volatile("s_wait_asynccnt " #n ::: "memory")
#endif

#if HAVE_ASYNC_LDS
// builtin prototype is (v4i addrspace(1)*, v4i addrspace(3)*, imm offset, imm cpol)
typedef int v4i_vs __attribute__((vector_size(16)));
typedef __attribute__((address_space(1))) v4i_vs* gv4i_p;
typedef __attribute__((address_space(3))) v4i_vs* lv4i_p;

// copy 16 bytes (8 halves) global -> LDS, async (ASYNCcnt)
__device__ __forceinline__ void async_cp16(const _Float16* g, _Float16* l) {
  __builtin_amdgcn_global_load_async_to_lds_b128(
      (gv4i_p)(v4i_vs*)(void*)g,
      (lv4i_p)(v4i_vs*)(void*)l, 0, 0);
}
#endif

// A-operand (16x32 f16) K index for vector element e, lane-half hi (contiguous: k = 16*hi + e)
__device__ __forceinline__ int a_kmap(int e, int hi) {
  return e + hi * 16;
}

// load A operand when K is contiguous in memory: ap points at row_base + k0 + hi*16
__device__ __forceinline__ v16h load_a_contig(const _Float16* ap) {
  AV u;
  *(H8*)(&u.h[0]) = *(const H8*)(ap);
  *(H8*)(&u.h[8]) = *(const H8*)(ap + 8);
  return u.v;
}

#define WMMA_F16(c, a, b) \
  (c) = __builtin_amdgcn_wmma_f32_16x16x32_f16(false, (a), false, (b), (short)0, (c), false, false)

// ---------------- small prep kernels ----------------

__global__ void k_cvt(const float* __restrict__ s, _Float16* __restrict__ d, int n) {
  int i = blockIdx.x * blockDim.x + threadIdx.x;
  if (i < n) d[i] = (_Float16)s[i];
}

// per-row sum of squares of x : x2[b*T+t]
__global__ void k_rowss(const float* __restrict__ x, float* __restrict__ x2) {
  __shared__ float sm[256];
  const float* row = x + (size_t)blockIdx.x * D_;
  float acc = 0.f;
  for (int d = threadIdx.x; d < D_; d += 256) { float v = row[d]; acc += v * v; }
  sm[threadIdx.x] = acc; __syncthreads();
  for (int off = 128; off > 0; off >>= 1) {
    if (threadIdx.x < off) sm[threadIdx.x] += sm[threadIdx.x + off];
    __syncthreads();
  }
  if (threadIdx.x == 0) x2[blockIdx.x] = sm[0];
}

// per-splat: ||p||^2, 0.5/scale^2, amplitude, and f16 copy of p
__global__ void k_splat(const float* __restrict__ p, const float* __restrict__ ls,
                        const float* __restrict__ am, _Float16* __restrict__ ph,
                        float* __restrict__ p2, float* __restrict__ inv2s2,
                        float* __restrict__ amp) {
  __shared__ float sm[256];
  int s = blockIdx.x;
  const float* row = p + s * D_;
  float acc = 0.f;
  for (int d = threadIdx.x; d < D_; d += 256) {
    float v = row[d];
    acc += v * v;
    ph[s * D_ + d] = (_Float16)v;
  }
  sm[threadIdx.x] = acc; __syncthreads();
  for (int off = 128; off > 0; off >>= 1) {
    if (threadIdx.x < off) sm[threadIdx.x] += sm[threadIdx.x + off];
    __syncthreads();
  }
  if (threadIdx.x == 0) {
    p2[s] = sm[0];
    inv2s2[s] = 0.5f * __expf(-2.f * ls[s]);   // 0.5 / exp(ls)^2
    amp[s] = am[s];
  }
}

// ---------------- WMMA GEMM kernels (one wave per 16x16 tile) ----------------

// Wh = f16( Wv @ Wo ), [D x D], K = D  (one-time 2 GFLOP; direct gathers)
__global__ void k_w(const float* __restrict__ Wv, const float* __restrict__ Wo,
                    _Float16* __restrict__ Wh) {
  const int lane = threadIdx.x & 31, l16 = lane & 15, hi = lane >> 4;
  const int tile = blockIdx.x;
  const int m0 = (tile / (D_ / 16)) * 16;
  const int n0 = (tile % (D_ / 16)) * 16;
  v8f c = {};
  for (int k0 = 0; k0 < D_; k0 += 32) {
    v16h a, b;
    const float* arow = Wv + (m0 + l16) * D_ + k0;
#pragma unroll
    for (int e = 0; e < 16; ++e) a[e] = (_Float16)arow[a_kmap(e, hi)];
#pragma unroll
    for (int e = 0; e < 16; ++e) b[e] = (_Float16)Wo[(k0 + e + hi * 16) * D_ + n0 + l16];
    WMMA_F16(c, a, b);
  }
#pragma unroll
  for (int r = 0; r < 8; ++r)
    Wh[(m0 + r + hi * 8) * D_ + n0 + l16] = (_Float16)c[r];
}

// xp tile via WMMA, then G = amp*exp(-max(x2+p2-2xp,0)*inv) -> Gh[B*T, S] (f16)
__global__ void k_gauss(const _Float16* __restrict__ xh, const _Float16* __restrict__ ph,
                        const float* __restrict__ x2, const float* __restrict__ p2,
                        const float* __restrict__ inv2s2, const float* __restrict__ amp,
                        _Float16* __restrict__ Gh) {
  const int lane = threadIdx.x & 31, l16 = lane & 15, hi = lane >> 4;
  const int m0 = blockIdx.x * 16;   // global row over B*T
  const int s0 = blockIdx.y * 16;   // splat tile
  v8f c = {};
  for (int k0 = 0; k0 < D_; k0 += 32) {
    v16h a = load_a_contig(xh + (m0 + l16) * D_ + k0 + hi * 16);
    v16h b = *(const v16h*)(ph + (s0 + l16) * D_ + k0 + hi * 16);   // contiguous K
    WMMA_F16(c, a, b);
  }
  const int s = s0 + l16;
  const float p2s = p2[s], invs = inv2s2[s], amps = amp[s];
#pragma unroll
  for (int r = 0; r < 8; ++r) {
    int row = m0 + r + hi * 8;
    float d2 = fmaxf(x2[row] + p2s - 2.f * c[r], 0.f);
    Gh[row * S_ + s] = (_Float16)(amps * __expf(-d2 * invs));
  }
}

// u[b,s] = sum_t G[b,t,s]
__global__ void k_u(const _Float16* __restrict__ Gh, float* __restrict__ u) {
  __shared__ float sm[256];
  int idx = blockIdx.x;            // b*S + s
  int b = idx / S_, s = idx % S_;
  float acc = 0.f;
  for (int t = threadIdx.x; t < T_; t += 256)
    acc += (float)Gh[(b * T_ + t) * S_ + s];
  sm[threadIdx.x] = acc; __syncthreads();
  for (int off = 128; off > 0; off >>= 1) {
    if (threadIdx.x < off) sm[threadIdx.x] += sm[threadIdx.x + off];
    __syncthreads();
  }
  if (threadIdx.x == 0) u[idx] = sm[0];
}

// Zh[b,s,d] = sum_t G[b,t,s] * x[b,t,d]   (A = G^T, B = x; both K-strided), K = T
// CDNA5 path: double-buffered GLOBAL_LOAD_ASYNC_TO_LDS staging of both 32xK tiles.
__global__ void k_z(const _Float16* __restrict__ Gh, const _Float16* __restrict__ xh,
                    _Float16* __restrict__ Zh) {
  const int lane = threadIdx.x & 31, l16 = lane & 15, hi = lane >> 4;
  int tile = blockIdx.x;           // b*256 + st*64 + dt
  int b = tile >> 8, rem = tile & 255;
  int sm0 = (rem >> 6) * 16;
  int n0 = (rem & 63) * 16;
  v8f c = {};
#if HAVE_ASYNC_LDS
  __shared__ alignas(16) _Float16 lA[2][32 * 16];
  __shared__ alignas(16) _Float16 lB[2][32 * 16];
  auto stage = [&](int buf, int k0) {
    // lane stages row t = k0+lane of both tiles (16 halves = 2x16B, coalesced)
    const _Float16* ga = Gh + (b * T_ + k0 + lane) * S_ + sm0;
    const _Float16* gb = xh + (b * T_ + k0 + lane) * D_ + n0;
    async_cp16(ga,     &lA[buf][lane * 16]);
    async_cp16(ga + 8, &lA[buf][lane * 16 + 8]);
    async_cp16(gb,     &lB[buf][lane * 16]);
    async_cp16(gb + 8, &lB[buf][lane * 16 + 8]);
  };
  stage(0, 0);
  int it = 0;
  for (int k0 = 0; k0 < T_; k0 += 32, ++it) {
    int cur = it & 1;
    if (k0 + 32 < T_) { stage(cur ^ 1, k0 + 32); ASYNC_WAIT(4); }
    else              { ASYNC_WAIT(0); }
    v16h a, bb;
#pragma unroll
    for (int e = 0; e < 16; ++e) a[e]  = lA[cur][a_kmap(e, hi) * 16 + l16];
#pragma unroll
    for (int e = 0; e < 16; ++e) bb[e] = lB[cur][(e + hi * 16) * 16 + l16];
    WMMA_F16(c, a, bb);
  }
#else
  for (int k0 = 0; k0 < T_; k0 += 32) {
    v16h a, bb;
#pragma unroll
    for (int e = 0; e < 16; ++e)
      a[e] = Gh[(b * T_ + k0 + a_kmap(e, hi)) * S_ + sm0 + l16];
#pragma unroll
    for (int e = 0; e < 16; ++e)
      bb[e] = xh[(b * T_ + k0 + e + hi * 16) * D_ + n0 + l16];
    WMMA_F16(c, a, bb);
  }
#endif
#pragma unroll
  for (int r = 0; r < 8; ++r)
    Zh[(b * S_ + sm0 + r + hi * 8) * D_ + n0 + l16] = (_Float16)c[r];
}

// Mh[b,s,e] = sum_d Z[b,s,d] * W[d,e]   (A contiguous; B staged via async LDS), K = D
__global__ void k_m(const _Float16* __restrict__ Zh, const _Float16* __restrict__ Wh,
                    _Float16* __restrict__ Mh) {
  const int lane = threadIdx.x & 31, l16 = lane & 15, hi = lane >> 4;
  int tile = blockIdx.x;
  int b = tile >> 8, rem = tile & 255;
  int sm0 = (rem >> 6) * 16;
  int n0 = (rem & 63) * 16;
  v8f c = {};
#if HAVE_ASYNC_LDS
  __shared__ alignas(16) _Float16 lB[2][32 * 16];
  auto stageB = [&](int buf, int k0) {
    const _Float16* gb = Wh + (k0 + lane) * D_ + n0;
    async_cp16(gb,     &lB[buf][lane * 16]);
    async_cp16(gb + 8, &lB[buf][lane * 16 + 8]);
  };
  stageB(0, 0);
  int it = 0;
  for (int k0 = 0; k0 < D_; k0 += 32, ++it) {
    int cur = it & 1;
    if (k0 + 32 < D_) { stageB(cur ^ 1, k0 + 32); ASYNC_WAIT(2); }
    else              { ASYNC_WAIT(0); }
    v16h a = load_a_contig(Zh + (b * S_ + sm0 + l16) * D_ + k0 + hi * 16);
    v16h bb;
#pragma unroll
    for (int e = 0; e < 16; ++e) bb[e] = lB[cur][(e + hi * 16) * 16 + l16];
    WMMA_F16(c, a, bb);
  }
#else
  for (int k0 = 0; k0 < D_; k0 += 32) {
    v16h a = load_a_contig(Zh + (b * S_ + sm0 + l16) * D_ + k0 + hi * 16);
    v16h bb;
#pragma unroll
    for (int e = 0; e < 16; ++e)
      bb[e] = Wh[(k0 + e + hi * 16) * D_ + n0 + l16];
    WMMA_F16(c, a, bb);
  }
#endif
#pragma unroll
  for (int r = 0; r < 8; ++r)
    Mh[(b * S_ + sm0 + r + hi * 8) * D_ + n0 + l16] = (_Float16)c[r];
}

// r[b,t] = sum_s G[b,t,s]*u[b,s] + EPS
__global__ void k_r(const _Float16* __restrict__ Gh, const float* __restrict__ u,
                    float* __restrict__ rr) {
  int row = blockIdx.x * blockDim.x + threadIdx.x;   // b*T + t
  if (row >= B_ * T_) return;
  int b = row / T_;
  float acc = EPS_;
  for (int s = 0; s < S_; ++s)
    acc += (float)Gh[row * S_ + s] * u[b * S_ + s];
  rr[row] = acc;
}

// out[b,t,e] = (sum_s G[b,t,s] * M[b,s,e]) / r[b,t]   (K = S = 64 -> 2 WMMA steps)
__global__ void k_out(const _Float16* __restrict__ Gh, const _Float16* __restrict__ Mh,
                      const float* __restrict__ rr, float* __restrict__ out) {
  const int lane = threadIdx.x & 31, l16 = lane & 15, hi = lane >> 4;
  int tile = blockIdx.x;           // b*8192 + mt*64 + nt
  int b = tile >> 13, rem = tile & 8191;
  int m0 = (rem >> 6) * 16;
  int n0 = (rem & 63) * 16;
  v8f c = {};
#if HAVE_ASYNC_LDS
  __shared__ alignas(16) _Float16 lB[2][32 * 16];
#pragma unroll
  for (int sstep = 0; sstep < 2; ++sstep) {
    const _Float16* gb = Mh + (b * S_ + sstep * 32 + lane) * D_ + n0;
    async_cp16(gb,     &lB[sstep][lane * 16]);
    async_cp16(gb + 8, &lB[sstep][lane * 16 + 8]);
  }
  ASYNC_WAIT(0);
#pragma unroll
  for (int sstep = 0; sstep < 2; ++sstep) {
    v16h a = load_a_contig(Gh + (b * T_ + m0 + l16) * S_ + sstep * 32 + hi * 16);
    v16h bb;
#pragma unroll
    for (int e = 0; e < 16; ++e) bb[e] = lB[sstep][(e + hi * 16) * 16 + l16];
    WMMA_F16(c, a, bb);
  }
#else
#pragma unroll
  for (int k0 = 0; k0 < S_; k0 += 32) {
    v16h a = load_a_contig(Gh + (b * T_ + m0 + l16) * S_ + k0 + hi * 16);
    v16h bb;
#pragma unroll
    for (int e = 0; e < 16; ++e)
      bb[e] = Mh[(b * S_ + k0 + e + hi * 16) * D_ + n0 + l16];
    WMMA_F16(c, a, bb);
  }
#endif
#pragma unroll
  for (int r = 0; r < 8; ++r) {
    int row = b * T_ + m0 + r + hi * 8;
    out[row * D_ + n0 + l16] = c[r] / rr[row];
  }
}

// ---------------- launcher ----------------

extern "C" void kernel_launch(void* const* d_in, const int* in_sizes, int n_in,
                              void* d_out, int out_size, void* d_ws, size_t ws_size,
                              hipStream_t stream) {
  (void)in_sizes; (void)n_in; (void)out_size; (void)ws_size;
  const float* x  = (const float*)d_in[0];   // [B,T,D]
  const float* sp = (const float*)d_in[1];   // [S,D]
  const float* ls = (const float*)d_in[2];   // [S]
  const float* am = (const float*)d_in[3];   // [S]
  const float* Wv = (const float*)d_in[4];   // [D,D]
  const float* Wo = (const float*)d_in[5];   // [D,D]
  float* out = (float*)d_out;                // [B,T,D]

  char* w = (char*)d_ws;
  size_t off = 0;
  auto alloc = [&](size_t bytes) -> void* {
    void* p = w + off;
    off += (bytes + 255) & ~(size_t)255;
    return p;
  };
  _Float16* xh = (_Float16*)alloc((size_t)B_ * T_ * D_ * 2);  // 16 MB
  _Float16* ph = (_Float16*)alloc((size_t)S_ * D_ * 2);       // 128 KB
  _Float16* Gh = (_Float16*)alloc((size_t)B_ * T_ * S_ * 2);  // 1 MB
  _Float16* Wh = (_Float16*)alloc((size_t)D_ * D_ * 2);       // 2 MB
  _Float16* Zh = (_Float16*)alloc((size_t)B_ * S_ * D_ * 2);  // 512 KB
  _Float16* Mh = (_Float16*)alloc((size_t)B_ * S_ * D_ * 2);  // 512 KB
  float* x2  = (float*)alloc((size_t)B_ * T_ * 4);
  float* p2  = (float*)alloc(S_ * 4);
  float* inv = (float*)alloc(S_ * 4);
  float* amp = (float*)alloc(S_ * 4);
  float* uu  = (float*)alloc((size_t)B_ * S_ * 4);
  float* rr  = (float*)alloc((size_t)B_ * T_ * 4);

  const int n = B_ * T_ * D_;
  k_cvt<<<(n + 255) / 256, 256, 0, stream>>>(x, xh, n);
  k_rowss<<<B_ * T_, 256, 0, stream>>>(x, x2);
  k_splat<<<S_, 256, 0, stream>>>(sp, ls, am, ph, p2, inv, amp);
  k_w<<<(D_ / 16) * (D_ / 16), 32, 0, stream>>>(Wv, Wo, Wh);
  dim3 gG(B_ * T_ / 16, S_ / 16);
  k_gauss<<<gG, 32, 0, stream>>>(xh, ph, x2, p2, inv, amp, Gh);
  k_u<<<B_ * S_, 256, 0, stream>>>(Gh, uu);
  k_z<<<B_ * (S_ / 16) * (D_ / 16), 32, 0, stream>>>(Gh, xh, Zh);
  k_m<<<B_ * (S_ / 16) * (D_ / 16), 32, 0, stream>>>(Zh, Wh, Mh);
  k_r<<<(B_ * T_ + 255) / 256, 256, 0, stream>>>(Gh, uu, rr);
  k_out<<<B_ * (T_ / 16) * (D_ / 16), 32, 0, stream>>>(Gh, Mh, rr, out);
}
